// MultiHeadAttention_54554674593949
// MI455X (gfx1250) — matrix operand order repaired and checked by reading, not compiled, not run on hardware
//
#include <hip/hip_runtime.h>
#include <cstdint>
#include <cstddef>

// ---------------------------------------------------------------------------
// MI455X (gfx1250, wave32) fused MHA forward.
// - All matmuls: v_wmma_f32_16x16x32_bf16 (f32 accumulation)
// - GEMM: weight k-slices (64x64, 8KB) async-staged to LDS via
//   global_load_async_to_lds_b128, double buffered, one s_wait_asynccnt +
//   barrier per 8 WMMAs. 32x32 wave tiles (2x fragment reuse).
// - Flash attention: softmax in log2 domain (scale folded into Q fragments),
//   full 64-key blocks without masking + <=2 masked 32-key diagonal blocks.
// - V GEMM epilogue transposes through LDS for contiguous stores.
// ---------------------------------------------------------------------------

typedef __bf16 bf16;
typedef __attribute__((ext_vector_type(16))) __bf16 v16bf;
typedef __attribute__((ext_vector_type(8)))  __bf16 v8bf;
typedef __attribute__((ext_vector_type(8)))  float  v8f;

static constexpr int Bc = 4, Sc = 2048, Dc = 1024, Hc = 16, HDc = 64;
static constexpr int MROWS = Bc * Sc; // 8192 token rows

#define WMMA_BF16(a, b, c)                                                  \
  __builtin_amdgcn_wmma_f32_16x16x32_bf16(false, (a), false, (b),           \
                                          (short)0, (c), false, false)

// ---- helpers ---------------------------------------------------------------

__device__ __forceinline__ bf16 f2bf(float f) {
  union { float f; uint32_t u; } v; v.f = f;
  uint32_t r = v.u + 0x7FFFu + ((v.u >> 16) & 1u); // round-to-nearest-even
  unsigned short h = (unsigned short)(r >> 16);
  return __builtin_bit_cast(bf16, h);
}
__device__ __forceinline__ float bf2f(bf16 b) {
  uint32_t u = ((uint32_t)__builtin_bit_cast(unsigned short, b)) << 16;
  return __builtin_bit_cast(float, u);
}

__device__ __forceinline__ v8f zero_v8f() {
  v8f z;
  for (int i = 0; i < 8; ++i) z[i] = 0.0f;
  return z;
}

// A-matrix fragment (16xK rows, 16-bit): lane l<16 holds row l, K {0..7,16..23};
// lane l>=16 holds row l-16, K {8..15, 24..31}.  (ISA 7.12.2, wave32)
__device__ __forceinline__ v16bf load_a_frag(const bf16* base, int lda, int lane) {
  const bf16* p = base + (size_t)(lane & 15) * lda + ((lane >> 4) << 3);
  v8bf lo = *(const v8bf*)(p);
  v8bf hi = *(const v8bf*)(p + 16);
  return __builtin_shufflevector(lo, hi, 0, 1, 2, 3, 4, 5, 6, 7,
                                 8, 9, 10, 11, 12, 13, 14, 15);
}

// B-matrix fragment (32x16, 16-bit): lane n holds column (n&15);
// lanes 0-15 -> K {0..7,16..23}, lanes 16-31 -> K {8..15,24..31}.
__device__ __forceinline__ v16bf load_b_frag(const bf16* base, int ldb, int lane) {
  const bf16* p = base + (size_t)(lane & 15) * ldb + ((lane >> 4) << 3);
  v8bf lo = *(const v8bf*)(p);
  v8bf hi = *(const v8bf*)(p + 16);
  return __builtin_shufflevector(lo, hi, 0, 1, 2, 3, 4, 5, 6, 7,
                                 8, 9, 10, 11, 12, 13, 14, 15);
}

// scale all 16 bf16 elements of a fragment by a float constant (done once)
__device__ __forceinline__ v16bf scale_frag(v16bf a, float s) {
  v16bf r;
  for (int i = 0; i < 16; ++i) r[i] = f2bf(bf2f(a[i]) * s);
  return r;
}

__device__ __forceinline__ float half_reduce_max(float v) {
  v = fmaxf(v, __shfl_xor(v, 1, 32));
  v = fmaxf(v, __shfl_xor(v, 2, 32));
  v = fmaxf(v, __shfl_xor(v, 4, 32));
  v = fmaxf(v, __shfl_xor(v, 8, 32));
  return v;
}
__device__ __forceinline__ float half_reduce_sum(float v) {
  v += __shfl_xor(v, 1, 32);
  v += __shfl_xor(v, 2, 32);
  v += __shfl_xor(v, 4, 32);
  v += __shfl_xor(v, 8, 32);
  return v;
}

// CDNA5 async copy: 16B global -> LDS, tracked with ASYNCcnt.
// GVS addressing: mem = SADDR(sgpr64) + VADDR[31:0]; LDS addr from VDST vgpr.
__device__ __forceinline__ void async_g2l_b128(uint32_t lds_byte_off,
                                               uint32_t g_byte_off,
                                               const void* g_base) {
  asm volatile("global_load_async_to_lds_b128 %0, %1, %2"
               :: "v"(lds_byte_off), "v"(g_byte_off), "s"(g_base)
               : "memory");
}
__device__ __forceinline__ void wait_asynccnt0() {
  asm volatile("s_wait_asynccnt 0x0" ::: "memory");
}

// ---- fp32 -> bf16 convert (128-bit streaming) ------------------------------

__global__ void cvt_f32_bf16(const float* __restrict__ in,
                             bf16* __restrict__ out, int n) {
  int i = (blockIdx.x * blockDim.x + threadIdx.x) * 8;
  if (i >= n) return;
  float4 a = *(const float4*)(in + i);
  float4 b = *(const float4*)(in + i + 4);
  v8bf o;
  o[0] = f2bf(a.x); o[1] = f2bf(a.y); o[2] = f2bf(a.z); o[3] = f2bf(a.w);
  o[4] = f2bf(b.x); o[5] = f2bf(b.y); o[6] = f2bf(b.z); o[7] = f2bf(b.w);
  *(v8bf*)(out + i) = o;
}

// ---- WMMA GEMM: C = A(MxK) * W^T, W is (NxK) row-major bf16 ----------------
// Block: 256 thr = 8 waves; block tile 128x64 (waves 4M x 2N, wave tile 32x32).
// Weight slice 64 rows x 64 k (8KB) async-staged, double buffered.

enum { MODE_QK = 0, MODE_V = 1, MODE_OUT = 2 };
static constexpr int KSTEP = 64;

template <int MODE>
__global__ __launch_bounds__(256) void gemm_wmma(
    const bf16* __restrict__ A, const bf16* __restrict__ W,
    void* __restrict__ Cout, const float* __restrict__ bias) {
  __shared__ __align__(16) bf16 wtile[2][64 * KSTEP];               // 2 x 8KB
  __shared__ __align__(16) bf16 vstore[(MODE == MODE_V) ? 8 * 32 * 40 : 1];

  const int tid  = threadIdx.x;
  const int lane = tid & 31;
  const int wid  = tid >> 5;
  const int tile_mb = blockIdx.y * 128;
  const int tile_nb = blockIdx.x * 64;
  const int r0 = tile_mb + (wid & 3) * 32;   // wave's 32 rows
  const int cN = (wid >> 2) * 32;            // wave's 32 cols (LDS-local)

  // async stage geometry: thread -> (row 0..63, 32B segment 0..3)
  const int srow = tid >> 2;
  const int scol = (tid & 3) << 4;  // element offset 0,16,32,48
  const uint32_t lds0 = (uint32_t)(uintptr_t)&wtile[0][0];
  const uint32_t lds_thr = lds0 + (uint32_t)(srow * KSTEP + scol) * 2u;
  const uint32_t g_thr = (uint32_t)((srow * Dc + scol) * 2);
  const bf16* wkbase = W + (size_t)tile_nb * Dc;

  v8f acc00 = zero_v8f(), acc01 = zero_v8f();
  v8f acc10 = zero_v8f(), acc11 = zero_v8f();

  const bf16* arow0 = A + (size_t)r0 * Dc;
  const bf16* arow1 = A + (size_t)(r0 + 16) * Dc;

  auto stage = [&](int buf, int k) {
    uint32_t dst = lds_thr + (uint32_t)buf * (uint32_t)(64 * KSTEP * 2);
    const bf16* base = wkbase + k;
    async_g2l_b128(dst,      g_thr,      base);
    async_g2l_b128(dst + 16, g_thr + 16, base);
  };

  stage(0, 0);  // prologue

  for (int k = 0; k < Dc; k += KSTEP) {
    const int cur = (k >> 6) & 1;
    wait_asynccnt0();   // own async copies for buffer `cur` complete
    __syncthreads();    // all waves' copies landed; prior reads finished
    if (k + KSTEP < Dc) stage(cur ^ 1, k + KSTEP);

    const bf16* wl = &wtile[cur][0];
#pragma unroll
    for (int kk = 0; kk < KSTEP; kk += 32) {
      v16bf a0 = load_a_frag(arow0 + k + kk, Dc, lane);
      v16bf a1 = load_a_frag(arow1 + k + kk, Dc, lane);
      v16bf b0 = load_b_frag(wl + (size_t)cN * KSTEP + kk, KSTEP, lane);
      v16bf b1 = load_b_frag(wl + (size_t)(cN + 16) * KSTEP + kk, KSTEP, lane);
      acc00 = WMMA_BF16(a0, b0, acc00);
      acc01 = WMMA_BF16(a0, b1, acc01);
      acc10 = WMMA_BF16(a1, b0, acc10);
      acc11 = WMMA_BF16(a1, b1, acc11);
    }
  }

  // C/D layout (ISA 7.12.2): VGPR i -> M = i + 8*(lane>=16), N = lane&15.
  const int half = lane >> 4;
  const int nlo  = lane & 15;

  if constexpr (MODE == MODE_V) {
    // transpose through LDS: lds[hd][s], then contiguous stores along s.
    bf16* vb = &vstore[wid * (32 * 40)];
#pragma unroll
    for (int ms = 0; ms < 2; ++ms)
#pragma unroll
      for (int ns = 0; ns < 2; ++ns) {
        v8f accv = (ms == 0) ? (ns == 0 ? acc00 : acc01)
                             : (ns == 0 ? acc10 : acc11);
#pragma unroll
        for (int i = 0; i < 8; ++i)
          vb[(ns * 16 + nlo) * 40 + ms * 16 + i + (half << 3)] = f2bf(accv[i]);
      }
    // wave-private region; DS ops are in-order per wave -> no barrier needed
    int h  = tile_nb >> 6;
    int hd = cN + lane;                 // this lane's Vt row
    int bi = r0 >> 11;
    int s0 = r0 & (Sc - 1);
    bf16* dst = (bf16*)Cout + (((size_t)(bi * Hc + h) * HDc) + hd) * Sc + s0;
    const bf16* src = vb + lane * 40;
#pragma unroll
    for (int seg = 0; seg < 4; ++seg)
      *(v8bf*)(dst + seg * 8) = *(const v8bf*)(src + seg * 8);
  } else {
#pragma unroll
    for (int ms = 0; ms < 2; ++ms)
#pragma unroll
      for (int ns = 0; ns < 2; ++ns) {
        v8f accv = (ms == 0) ? (ns == 0 ? acc00 : acc01)
                             : (ns == 0 ? acc10 : acc11);
#pragma unroll
        for (int i = 0; i < 8; ++i) {
          int m = r0 + ms * 16 + i + (half << 3);
          int bi = m >> 11;
          int s  = m & (Sc - 1);
          int n = tile_nb + cN + ns * 16 + nlo;
          float vv = accv[i];
          if constexpr (MODE == MODE_QK) {
            int h = n >> 6, hd = n & 63;   // Q/K: [B,H,S,HD] bf16
            ((bf16*)Cout)[(((size_t)(bi * Hc + h) * Sc) + s) * HDc + hd] = f2bf(vv);
          } else {                          // out-proj: fp32 + bias
            ((float*)Cout)[(size_t)m * Dc + n] = vv + bias[n];
          }
        }
      }
  }
}

// ---- Flash attention: 1 wave = 16 query rows --------------------------------
// Full 64-key blocks (no mask) + <=2 masked 32-key diagonal blocks.
// Q fragments pre-scaled by (1/sqrt(HD))*log2(e): softmax in log2 domain.

__global__ __launch_bounds__(128) void attn_flash(
    const bf16* __restrict__ Q, const bf16* __restrict__ K,
    const bf16* __restrict__ Vt, bf16* __restrict__ ctx) {
  __shared__ __align__(16) bf16 pbuf[4][16 * 64];  // per-wave P staging, 8KB

  const int lane = threadIdx.x & 31;
  const int w    = threadIdx.x >> 5;
  const int h = blockIdx.y, b = blockIdx.z;
  const int q0 = blockIdx.x * 64 + w * 16;

  const bf16* Qb = Q  + ((size_t)(b * Hc + h) * Sc) * HDc;
  const bf16* Kb = K  + ((size_t)(b * Hc + h) * Sc) * HDc;
  const bf16* Vb = Vt + ((size_t)(b * Hc + h) * HDc) * Sc;

  const float kQScale = 0.125f * 1.44269504088896f;  // 1/sqrt(64) * log2(e)
  v16bf aq0 = scale_frag(load_a_frag(Qb + (size_t)q0 * HDc,      HDc, lane), kQScale);
  v16bf aq1 = scale_frag(load_a_frag(Qb + (size_t)q0 * HDc + 32, HDc, lane), kQScale);

  const int half = lane >> 4;
  const int nlo  = lane & 15;

  float mrow[8], lrow[8];
  v8f acc[4];
  for (int i = 0; i < 8; ++i) { mrow[i] = -1e30f; lrow[i] = 0.0f; }
  for (int t = 0; t < 4; ++t) acc[t] = zero_v8f();

  bf16* pb = &pbuf[w][0];
  const int qmax = q0 + 15;
  const int nfull = q0 >> 6;  // 64-key blocks fully below the diagonal

  int c = 0;
  for (int fb = 0; fb < nfull; ++fb, c += 64) {
    // ---- scores: 4 tiles of 16 keys, log2 domain, no masking ----
    v8f st[4];
#pragma unroll
    for (int t = 0; t < 4; ++t) {
      st[t] = zero_v8f();
      st[t] = WMMA_BF16(aq0, load_b_frag(Kb + (size_t)(c + t * 16) * HDc,      HDc, lane), st[t]);
      st[t] = WMMA_BF16(aq1, load_b_frag(Kb + (size_t)(c + t * 16) * HDc + 32, HDc, lane), st[t]);
    }
#pragma unroll
    for (int i = 0; i < 8; ++i) {
      float bm = half_reduce_max(fmaxf(fmaxf(st[0][i], st[1][i]),
                                       fmaxf(st[2][i], st[3][i])));
      float mnew = fmaxf(mrow[i], bm);
      float alpha = exp2f(mrow[i] - mnew);
      float e0 = exp2f(st[0][i] - mnew);
      float e1 = exp2f(st[1][i] - mnew);
      float e2 = exp2f(st[2][i] - mnew);
      float e3 = exp2f(st[3][i] - mnew);
      float rs = half_reduce_sum((e0 + e1) + (e2 + e3));
      lrow[i] = lrow[i] * alpha + rs;
      mrow[i] = mnew;
      acc[0][i] *= alpha; acc[1][i] *= alpha;
      acc[2][i] *= alpha; acc[3][i] *= alpha;
      int r = i + (half << 3);
      pb[r * 64 + nlo]      = f2bf(e0);
      pb[r * 64 + 16 + nlo] = f2bf(e1);
      pb[r * 64 + 32 + nlo] = f2bf(e2);
      pb[r * 64 + 48 + nlo] = f2bf(e3);
    }
    v16bf pa0 = load_a_frag(pb,      64, lane);
    v16bf pa1 = load_a_frag(pb + 32, 64, lane);
#pragma unroll
    for (int t = 0; t < 4; ++t) {
      acc[t] = WMMA_BF16(pa0, load_b_frag(Vb + (size_t)(t * 16) * Sc + c,      Sc, lane), acc[t]);
      acc[t] = WMMA_BF16(pa1, load_b_frag(Vb + (size_t)(t * 16) * Sc + c + 32, Sc, lane), acc[t]);
    }
  }

  // ---- masked diagonal blocks: 32 keys each ----
  for (; c <= qmax; c += 32) {
    v8f s0 = zero_v8f(), s1 = zero_v8f();
    s0 = WMMA_BF16(aq0, load_b_frag(Kb + (size_t)c * HDc,             HDc, lane), s0);
    s0 = WMMA_BF16(aq1, load_b_frag(Kb + (size_t)c * HDc + 32,        HDc, lane), s0);
    s1 = WMMA_BF16(aq0, load_b_frag(Kb + (size_t)(c + 16) * HDc,      HDc, lane), s1);
    s1 = WMMA_BF16(aq1, load_b_frag(Kb + (size_t)(c + 16) * HDc + 32, HDc, lane), s1);
#pragma unroll
    for (int i = 0; i < 8; ++i) {
      int q = q0 + i + (half << 3);
      float v0 = ((c + nlo)      <= q) ? s0[i] : -1e30f;
      float v1 = ((c + 16 + nlo) <= q) ? s1[i] : -1e30f;
      float bm = half_reduce_max(fmaxf(v0, v1));
      float mnew = fmaxf(mrow[i], bm);
      float alpha = exp2f(mrow[i] - mnew);
      float e0 = exp2f(v0 - mnew);
      float e1 = exp2f(v1 - mnew);
      float rs = half_reduce_sum(e0 + e1);
      lrow[i] = lrow[i] * alpha + rs;
      mrow[i] = mnew;
      acc[0][i] *= alpha; acc[1][i] *= alpha;
      acc[2][i] *= alpha; acc[3][i] *= alpha;
      int r = i + (half << 3);
      pb[r * 64 + nlo]      = f2bf(e0);
      pb[r * 64 + 16 + nlo] = f2bf(e1);
    }
    v16bf pa = load_a_frag(pb, 64, lane);
#pragma unroll
    for (int t = 0; t < 4; ++t) {
      v16bf bv = load_b_frag(Vb + (size_t)(t * 16) * Sc + c, Sc, lane);
      acc[t] = WMMA_BF16(pa, bv, acc[t]);
    }
  }

  // normalize + store ctx bf16 row-major [B*S, D]
  for (int i = 0; i < 8; ++i) {
    int q = q0 + i + (half << 3);
    float inv = 1.0f / lrow[i];
    size_t rowoff = ((size_t)(b * Sc + q)) * Dc + h * HDc;
#pragma unroll
    for (int t = 0; t < 4; ++t)
      ctx[rowoff + t * 16 + nlo] = f2bf(acc[t][i] * inv);
  }
}

// ---- host launch -----------------------------------------------------------

extern "C" void kernel_launch(void* const* d_in, const int* in_sizes, int n_in,
                              void* d_out, int out_size, void* d_ws, size_t ws_size,
                              hipStream_t stream) {
  const float* x   = (const float*)d_in[0];
  const float* W_q = (const float*)d_in[1];
  const float* W_k = (const float*)d_in[2];
  const float* W_v = (const float*)d_in[3];
  const float* W_o = (const float*)d_in[4];
  const float* b_o = (const float*)d_in[5];

  char* ws = (char*)d_ws;
  size_t off = 0;
  auto alloc = [&](size_t bytes) -> void* {
    void* p = ws + off;
    off += (bytes + 255) & ~(size_t)255;
    return p;
  };
  bf16* xb  = (bf16*)alloc((size_t)MROWS * Dc * sizeof(bf16)); // 16 MB
  bf16* wq  = (bf16*)alloc((size_t)Dc * Dc * sizeof(bf16));    //  2 MB
  bf16* wk  = (bf16*)alloc((size_t)Dc * Dc * sizeof(bf16));
  bf16* wv  = (bf16*)alloc((size_t)Dc * Dc * sizeof(bf16));
  bf16* wo  = (bf16*)alloc((size_t)Dc * Dc * sizeof(bf16));
  bf16* Qb  = (bf16*)alloc((size_t)MROWS * Dc * sizeof(bf16)); // 16 MB
  bf16* Kb  = (bf16*)alloc((size_t)MROWS * Dc * sizeof(bf16));
  bf16* Vt  = (bf16*)alloc((size_t)MROWS * Dc * sizeof(bf16));
  bf16* ctx = (bf16*)alloc((size_t)MROWS * Dc * sizeof(bf16));
  (void)ws_size; (void)in_sizes; (void)n_in; (void)out_size;

  const int NX = MROWS * Dc;  // 8M
  const int NW = Dc * Dc;     // 1M
  cvt_f32_bf16<<<(NX / 8 + 255) / 256, 256, 0, stream>>>(x, xb, NX);
  cvt_f32_bf16<<<(NW / 8 + 255) / 256, 256, 0, stream>>>(W_q, wq, NW);
  cvt_f32_bf16<<<(NW / 8 + 255) / 256, 256, 0, stream>>>(W_k, wk, NW);
  cvt_f32_bf16<<<(NW / 8 + 255) / 256, 256, 0, stream>>>(W_v, wv, NW);
  cvt_f32_bf16<<<(NW / 8 + 255) / 256, 256, 0, stream>>>(W_o, wo, NW);

  dim3 gg(Dc / 64, MROWS / 128);  // 16 x 64 blocks, 256 thr each
  gemm_wmma<MODE_QK><<<gg, 256, 0, stream>>>(xb, wq, Qb, nullptr);
  gemm_wmma<MODE_QK><<<gg, 256, 0, stream>>>(xb, wk, Kb, nullptr);
  gemm_wmma<MODE_V ><<<gg, 256, 0, stream>>>(xb, wv, Vt, nullptr);

  attn_flash<<<dim3(Sc / 64, Hc, Bc), 128, 0, stream>>>(Qb, Kb, Vt, ctx);

  gemm_wmma<MODE_OUT><<<gg, 256, 0, stream>>>(ctx, wo, d_out, b_o);
}
